// DeformableConv2d_V2_52999896432824
// MI455X (gfx1250) — matrix operand order repaired and checked by reading, not compiled, not run on hardware
//
#include <hip/hip_runtime.h>

typedef __attribute__((ext_vector_type(16))) __bf16          v16bf;
typedef __attribute__((ext_vector_type(8)))  float           v8f;
typedef __attribute__((ext_vector_type(8)))  unsigned short  u16x8;
typedef int v4i __attribute__((vector_size(16)));

#define C_IN   512
#define HH     96
#define WW     96
#define BB     2
#define GG     8
#define K2C    9
#define CG     64            // C/G
#define SPC    (HH*WW)       // 9216
#define NPIX   (BB*SPC)      // 18432
#define KG1    (C_IN*K2C)    // 4608  (K for offset/mask convs, tap-major)
#define CO_OFF 144
#define CO_MSK 72
#define KG2    (CG*K2C)      // 576   (K for grouped main conv)
#define BPAD   8             // LDS row pad (ushorts) -> conflict-free b128 rows

__device__ __forceinline__ unsigned short f2bf_bits(float f) {
  unsigned int u = __builtin_bit_cast(unsigned int, f);
  unsigned int r = u + 0x7FFFu + ((u >> 16) & 1u);   // round-to-nearest-even
  return (unsigned short)(r >> 16);
}
__device__ __forceinline__ __bf16 bits2bf(unsigned short s) {
  return __builtin_bit_cast(__bf16, s);
}
// Assemble a WMMA 16-bit fragment from two contiguous 8-element chunks.
__device__ __forceinline__ v16bf mk16(u16x8 lo, u16x8 hi) {
  v16bf r;
#pragma unroll
  for (int e = 0; e < 8; ++e) {
    r[e]     = bits2bf(lo[e]);
    r[e + 8] = bits2bf(hi[e]);
  }
  return r;
}

// ---------------------------------------------------------------- helpers
__global__ __launch_bounds__(256)
void zero_u16(unsigned short* __restrict__ p, int n) {
  int i = blockIdx.x * 256 + threadIdx.x;
  if (i < n) p[i] = 0;
}

__global__ __launch_bounds__(256)
void cvt_bf16(const float* __restrict__ src, unsigned short* __restrict__ dst, int n) {
  int i = blockIdx.x * blockDim.x + threadIdx.x;
  if (i < n) dst[i] = f2bf_bits(src[i]);
}

// ------------------------------------------- feat NCHW fp32 -> NHWC bf16
__global__ __launch_bounds__(256)
void repack_feat_nhwc(const float* __restrict__ src, unsigned short* __restrict__ dst) {
  long i = (long)blockIdx.x * 256 + threadIdx.x;     // over B*H*W*C
  if (i >= (long)BB * SPC * C_IN) return;
  int  ci = (int)(i & (C_IN - 1));
  long p  = i >> 9;                                   // b*SPC + sp
  int  b  = (int)(p / SPC);
  int  sp = (int)(p - (long)b * SPC);
  dst[i] = f2bf_bits(src[((long)b * C_IN + ci) * SPC + sp]);
}

// ---------------------- conv weights (co,ci,kh,kw) fp32 -> (co,tap,ci) bf16
__global__ __launch_bounds__(256)
void repack_w_tapmajor(const float* __restrict__ src, unsigned short* __restrict__ dst,
                       int Co) {
  int i = blockIdx.x * 256 + threadIdx.x;            // over Co*4608
  if (i >= Co * KG1) return;
  int co  = i / KG1;
  int r   = i - co * KG1;
  int tap = r >> 9;
  int ci  = r & (C_IN - 1);
  dst[i] = f2bf_bits(src[(long)co * KG1 + (long)ci * K2C + tap]);
}

// --------------------------------------------- implicit-GEMM 3x3 conv via WMMA
// Block = (32, nCoTiles) threads: each wave owns one 16-row Co tile; all waves
// share one 32px x 512ch feat tile per tap, staged in LDS by async DMA.
__global__ __launch_bounds__(288)
void conv3x3_wmma(const unsigned short* __restrict__ featn,   // NHWC bf16
                  const unsigned short* __restrict__ wb,      // (co, tap, ci) bf16
                  const unsigned short* __restrict__ zbuf,    // zeroed, >=1024
                  const float* __restrict__ bias,
                  float* __restrict__ out, int Co, int do_sigmoid) {
  __shared__ alignas(16) unsigned short btile[32][C_IN + BPAD];  // ~33 KB

  const int nth  = blockDim.x * blockDim.y;
  const int t    = threadIdx.y * 32 + threadIdx.x;
  const int lane = threadIdx.x;
  const int half = lane >> 4;
  const int mn   = lane & 15;

  // 32-pixel tile lies in a single image row (96 % 32 == 0)
  const int pbase = blockIdx.x * 32;
  const int b0 = pbase / SPC;
  const int sp0 = pbase - b0 * SPC;
  const int h0 = sp0 / WW;
  const int w0 = sp0 - h0 * WW;

  const int coA  = threadIdx.y * 16 + mn;
  const int coAc = coA < Co ? coA : (Co - 1);
  const unsigned short* wrow = wb + (long)coAc * KG1;

  v8f acc0 = {}, acc1 = {};

  for (int tap = 0; tap < K2C; ++tap) {
    const int kh = tap / 3, kw = tap - kh * 3;
    const int y  = h0 + kh - 1;
    const bool okY = (unsigned)y < (unsigned)HH;
    const unsigned short* srow =
        featn + ((long)(b0 * HH + y) * WW + (w0 + kw - 1)) * C_IN;

    // cooperative async stage of the 32px x 512ch B tile (zero rows via zbuf)
    for (int i = t; i < 32 * (C_IN / 8); i += nth) {
      int px  = i >> 6;
      int cof = (i & 63) * 8;
      int xx  = w0 + px + kw - 1;
      const unsigned short* gsrc =
          (okY && (unsigned)xx < (unsigned)WW) ? (srow + (long)px * C_IN + cof)
                                               : (zbuf + cof);
#if __has_builtin(__builtin_amdgcn_global_load_async_to_lds_b128)
      __builtin_amdgcn_global_load_async_to_lds_b128(
          (v4i*)gsrc, (v4i*)&btile[px][cof], 0, 0);
#else
      *(u16x8*)&btile[px][cof] = *(const u16x8*)gsrc;
#endif
    }
#if __has_builtin(__builtin_amdgcn_s_wait_asynccnt)
    __builtin_amdgcn_s_wait_asynccnt(0);
#else
    asm volatile("s_wait_asynccnt 0" ::: "memory");
#endif
    __syncthreads();

    const unsigned short* wt = wrow + tap * C_IN;
    for (int cc = 0; cc < C_IN; cc += 32) {
      __builtin_prefetch((const void*)(wt + cc + 128), 0, 1);
      // A: two contiguous 16B chunks, k = 16*half + s (matches B)
      u16x8 aLo = *(const u16x8*)(wt + cc + half * 16);
      u16x8 aHi = *(const u16x8*)(wt + cc + half * 16 + 8);
      v16bf a = mk16(aLo, aHi);
      // B from LDS: two pixels (columns) per lane-pair of WMMAs
      u16x8 b0lo = *(const u16x8*)(&btile[mn][cc + half * 16]);
      u16x8 b0hi = *(const u16x8*)(&btile[mn][cc + half * 16 + 8]);
      u16x8 b1lo = *(const u16x8*)(&btile[mn + 16][cc + half * 16]);
      u16x8 b1hi = *(const u16x8*)(&btile[mn + 16][cc + half * 16 + 8]);
      acc0 = __builtin_amdgcn_wmma_f32_16x16x32_bf16(false, a, false, mk16(b0lo, b0hi),
                                                     (short)0, acc0, false, false);
      acc1 = __builtin_amdgcn_wmma_f32_16x16x32_bf16(false, a, false, mk16(b1lo, b1hi),
                                                     (short)0, acc1, false, false);
    }
    __syncthreads();
  }

#pragma unroll
  for (int r = 0; r < 8; ++r) {
    int M  = r + half * 8;
    int co = threadIdx.y * 16 + M;
    if (co < Co) {
      float v0 = acc0[r] + bias[co];
      float v1 = acc1[r] + bias[co];
      if (do_sigmoid) {
        v0 = 1.0f / (1.0f + __expf(-v0));
        v1 = 1.0f / (1.0f + __expf(-v1));
      }
      long base = ((long)b0 * Co + co) * SPC + sp0 + mn;
      out[base]      = v0;
      out[base + 16] = v1;
    }
  }
}

// ---------------------------------------- deformable sampling + grouped GEMM
// Block: 128 threads (4 waves).  Grid: (NPIX/16, G).
__global__ __launch_bounds__(128)
void dcn_wmma(const float* __restrict__ x,
              const float* __restrict__ offs,
              const float* __restrict__ msk,
              const unsigned short* __restrict__ mwb,   // (co, c*9+k) bf16
              const float* __restrict__ bias,
              float* __restrict__ out) {
  // transposed sample tile: samp[pixel][ck], +8 pad -> conflict-free b128 rows
  __shared__ alignas(16) unsigned short samp[16][KG2 + BPAD];
  __shared__ float cw[4][144];
  __shared__ int   cidx[4][144];

  const int g     = blockIdx.y;
  const int pbase = blockIdx.x * 16;
  const int b0    = pbase / SPC;
  const int t     = threadIdx.x;

  // ---- stage 1: bilinear corner indices + weights (mask folded), 9 taps x 16 px
  for (int c = t; c < 144; c += 128) {
    int k = c >> 4;
    int n = c & 15;
    int p  = pbase + n;
    int sp = p - b0 * SPC;
    int h  = sp / WW;
    int w  = sp - h * WW;
    int och = (g * K2C + k) * 2;
    float oy = offs[((long)b0 * (2 * GG * K2C) + och)     * SPC + sp];
    float ox = offs[((long)b0 * (2 * GG * K2C) + och + 1) * SPC + sp];
    float m  = msk [((long)b0 * (GG * K2C) + g * K2C + k) * SPC + sp];
    float py = (float)(h + k / 3 - 1) + oy;
    float px = (float)(w + k % 3 - 1) + ox;
    float y0f = floorf(py), x0f = floorf(px);
    float ly = py - y0f, lx = px - x0f;
    int y0 = (int)y0f, x0 = (int)x0f;
    cw[0][c] = (1.f - ly) * (1.f - lx) * m;
    cw[1][c] = (1.f - ly) * lx * m;
    cw[2][c] = ly * (1.f - lx) * m;
    cw[3][c] = ly * lx * m;
    cidx[0][c] = ((unsigned)y0       < (unsigned)HH && (unsigned)x0       < (unsigned)WW) ?  y0      * WW + x0     : -1;
    cidx[1][c] = ((unsigned)y0       < (unsigned)HH && (unsigned)(x0 + 1) < (unsigned)WW) ?  y0      * WW + x0 + 1 : -1;
    cidx[2][c] = ((unsigned)(y0 + 1) < (unsigned)HH && (unsigned)x0       < (unsigned)WW) ? (y0 + 1) * WW + x0     : -1;
    cidx[3][c] = ((unsigned)(y0 + 1) < (unsigned)HH && (unsigned)(x0 + 1) < (unsigned)WW) ? (y0 + 1) * WW + x0 + 1 : -1;
  }
  __syncthreads();

  // ---- stage 2: build transposed 16 x 576 bf16 sample tile
  const long xbase_g = ((long)b0 * C_IN + g * CG) * SPC;
  for (int e2 = t; e2 < KG2 * 16; e2 += 128) {
    int ck = e2 >> 4;
    int n  = e2 & 15;
    int c  = ck / K2C;
    int k  = ck - c * K2C;
    int combo = (k << 4) | n;
    const float* xc = x + xbase_g + (long)c * SPC;
    float v = 0.f;
#pragma unroll
    for (int q = 0; q < 4; ++q) {
      int si = cidx[q][combo];
      if (si >= 0) v += xc[si] * cw[q][combo];
    }
    samp[n][ck] = f2bf_bits(v);
  }
  __syncthreads();

  // ---- stage 3: 4 waves, each one 16-row output slab; 18-step bf16 WMMA loop
  const int wid  = t >> 5;
  const int lane = t & 31;
  const int half = lane >> 4;
  const int mn   = lane & 15;
  const unsigned short* arow = mwb + (long)(g * CG + wid * 16 + mn) * KG2;

  v8f acc = {};
  for (int kb = 0; kb < KG2; kb += 32) {
    u16x8 aLo = *(const u16x8*)(arow + kb + half * 16);
    u16x8 aHi = *(const u16x8*)(arow + kb + half * 16 + 8);
    u16x8 bLo = *(const u16x8*)(&samp[mn][kb + half * 16]);
    u16x8 bHi = *(const u16x8*)(&samp[mn][kb + half * 16 + 8]);
    v16bf a  = mk16(aLo, aHi);
    v16bf bm = mk16(bLo, bHi);
    acc = __builtin_amdgcn_wmma_f32_16x16x32_bf16(false, a, false, bm,
                                                  (short)0, acc, false, false);
  }

  const int p  = pbase + mn;
  const int sp = p - b0 * SPC;
#pragma unroll
  for (int r = 0; r < 8; ++r) {
    int co = g * CG + wid * 16 + r + half * 8;
    out[((long)b0 * C_IN + co) * SPC + sp] = (acc[r] + bias[co]) * 2.0f;
  }
}

// ---------------------------------------------------------------------- launch
extern "C" void kernel_launch(void* const* d_in, const int* in_sizes, int n_in,
                              void* d_out, int out_size, void* d_ws, size_t ws_size,
                              hipStream_t stream) {
  const float* x     = (const float*)d_in[0];
  const float* feat  = (const float*)d_in[1];
  const float* offw  = (const float*)d_in[2];
  const float* offb  = (const float*)d_in[3];
  const float* maskw = (const float*)d_in[4];
  const float* maskb = (const float*)d_in[5];
  const float* mainw = (const float*)d_in[6];
  const float* mainb = (const float*)d_in[7];

  float* out_main = (float*)d_out;                            // B*C*H*W
  float* out_off  = out_main + (long)BB * C_IN * SPC;         // B*144*H*W
  float* out_mask = out_off  + (long)BB * 2 * GG * K2C * SPC; // B*72*H*W

  char* ws = (char*)d_ws;
  auto takeb = [&](long elems) {
    unsigned short* ptr = (unsigned short*)ws;
    ws += ((elems * 2 + 255) & ~255L);
    return ptr;
  };
  unsigned short* featn  = takeb((long)BB * SPC * C_IN);   // NHWC bf16
  unsigned short* offwb  = takeb((long)CO_OFF * KG1);      // tap-major
  unsigned short* maskwb = takeb((long)CO_MSK * KG1);      // tap-major
  unsigned short* mainwb = takeb((long)C_IN * KG2);        // (co, ck)
  unsigned short* zbuf   = takeb(1024);                    // zero page

  zero_u16<<<4, 256, 0, stream>>>(zbuf, 1024);
  {
    long n = (long)BB * SPC * C_IN;
    repack_feat_nhwc<<<(int)((n + 255) / 256), 256, 0, stream>>>(feat, featn);
  }
  repack_w_tapmajor<<<(CO_OFF * KG1 + 255) / 256, 256, 0, stream>>>(offw,  offwb,  CO_OFF);
  repack_w_tapmajor<<<(CO_MSK * KG1 + 255) / 256, 256, 0, stream>>>(maskw, maskwb, CO_MSK);
  cvt_bf16<<<(C_IN * KG2 + 255) / 256, 256, 0, stream>>>(mainw, mainwb, C_IN * KG2);

  // offset conv: 9 Co-tiles/block; mask conv: 5 Co-tiles/block (+sigmoid)
  conv3x3_wmma<<<dim3(NPIX / 32), dim3(32, (CO_OFF + 15) / 16), 0, stream>>>(
      featn, offwb, zbuf, offb, out_off, CO_OFF, 0);
  conv3x3_wmma<<<dim3(NPIX / 32), dim3(32, (CO_MSK + 15) / 16), 0, stream>>>(
      featn, maskwb, zbuf, maskb, out_mask, CO_MSK, 1);

  dcn_wmma<<<dim3(NPIX / 16, GG), 128, 0, stream>>>(
      x, out_off, out_mask, mainwb, mainb, out_main);
}
